// TempoJelly_15384572854746
// MI455X (gfx1250) — matrix operand order repaired and checked by reading, not compiled, not run on hardware
//
#include <hip/hip_runtime.h>
#include <hip/hip_bf16.h>

// ---------------------------------------------------------------------------
// SNN 2-layer forward: s2 = ((((x @ W1^T) >= 1) @ W2^T) >= 1)
// B=4096, IN=3136, FEAT=6272, OUT=500
// fp32 -> f16 staging, then two WMMA f16 (f32-accum) NT GEMMs.
// Staging: async global->LDS (ASYNCcnt), 3-deep LDS pipeline with peeled
// drain so the hot loop is branch-free; coalesced f16 epilogue via reused LDS.
// Fallback (toolchains without async builtins): register-pipelined loads.
// ---------------------------------------------------------------------------

typedef __attribute__((ext_vector_type(16))) _Float16 v16h;
typedef __attribute__((ext_vector_type(8)))  _Float16 v8h;   // 16 bytes
typedef __attribute__((ext_vector_type(4)))  _Float16 v4h;   // 8 bytes
typedef __attribute__((ext_vector_type(8)))  float    v8f;
typedef __attribute__((ext_vector_type(4)))  int      v4i;   // 16 bytes

union F16Frag { v16h v; v8h h[2]; };

#if defined(__has_builtin)
#  if __has_builtin(__builtin_amdgcn_global_load_async_to_lds_b128) && \
      __has_builtin(__builtin_amdgcn_s_wait_asynccnt)
#    define USE_ASYNC_LDS 1
#  endif
#endif
#ifndef USE_ASYNC_LDS
#  define USE_ASYNC_LDS 0
#endif

#define AS1 __attribute__((address_space(1)))
#define AS3 __attribute__((address_space(3)))

#if USE_ASYNC_LDS
__device__ __forceinline__ void async_cp_b128(const _Float16* gsrc, _Float16* ldst) {
  // global_load_async_to_lds_b128 (ASYNCcnt), probe-confirmed prototype:
  // (AS1 v4i* gsrc, AS3 v4i* ldst, imm offset, imm cpol).
  __builtin_amdgcn_global_load_async_to_lds_b128(
      (AS1 v4i*)(AS1 void*)(void*)gsrc,
      (AS3 v4i*)(AS3 void*)(void*)ldst, 0, 0);
}
#endif

// ---------------------------------------------------------------------------
// fp32 -> f16 elementwise conversion (vectorized x4)
// ---------------------------------------------------------------------------
__global__ __launch_bounds__(256) void cvt_f32_to_f16(
    const float* __restrict__ in, _Float16* __restrict__ out, int n4) {
  int i = blockIdx.x * blockDim.x + threadIdx.x;
  if (i < n4) {
    float4 v = ((const float4*)in)[i];
    v4h o;
    o[0] = (_Float16)v.x; o[1] = (_Float16)v.y;
    o[2] = (_Float16)v.z; o[3] = (_Float16)v.w;
    ((v4h*)out)[i] = o;
  }
}

// fp32 -> f16 with row padding: out is [Npad][K], rows >= Nin are zero.
__global__ __launch_bounds__(256) void cvt_pad_f32_to_f16(
    const float* __restrict__ in, _Float16* __restrict__ out,
    int K, int Nin, int Npad) {
  int i = blockIdx.x * blockDim.x + threadIdx.x;
  int total4 = (Npad * K) >> 2;
  if (i < total4) {
    int e = i << 2;
    int n = e / K;           // K is a multiple of 4 -> no row crossing
    int k = e - n * K;
    v4h o;
    if (n < Nin) {
      float4 v = *(const float4*)&in[(size_t)n * K + k];
      o[0] = (_Float16)v.x; o[1] = (_Float16)v.y;
      o[2] = (_Float16)v.z; o[3] = (_Float16)v.w;
    } else {
      o[0] = o[1] = o[2] = o[3] = (_Float16)0.0f;
    }
    ((v4h*)out)[i] = o;
  }
}

// ---------------------------------------------------------------------------
// NT GEMM + threshold:  C[m][n] = sum_k A[m][k] * W[n][k];  out = (C >= 1)
//   A: [M][K] f16 row-major, W: [N][K] f16 row-major (K % 64 == 0)
//   STORE_F16: out f16 0/1 at [M][N].  else: out f32 0/1 at [M][Nstore], col guard.
// Tile: 128x128 per workgroup, BK=64; 8 waves in 4(M) x 2(N), each wave 32x64.
// ---------------------------------------------------------------------------
#define BK 64
#define LDSW (BK + 8)   // 72 halves: 144B row stride -> conflict-free b128 reads
#define OSTR 136        // epilogue LDS tile row stride (halves), conflict-free

template <bool STORE_F16>
__global__ __launch_bounds__(256) void snn_gemm_nt(
    const _Float16* __restrict__ A,
    const _Float16* __restrict__ W,
    void* __restrict__ Cout,
    int M, int N, int K, int Nstore) {
  (void)M;
  const int tid   = threadIdx.x;
  const int lane  = tid & 31;          // wave32
  const int wave  = tid >> 5;          // 0..7
  const int wm    = wave >> 1;         // 0..3 : 32-row slice
  const int wn    = wave & 1;          // 0..1 : 64-col slice
  const int m0    = blockIdx.x * 128;
  const int n0    = blockIdx.y * 128;
  const int l16   = lane & 15;
  const int lhalf = lane >> 4;         // 0 or 1

  v8f acc[2][4] = {};

  // 2 K-steps of 32, 8 WMMAs each, reading one (A,B) tile pair from LDS.
  auto compute_tile = [&](const _Float16* lA, const _Float16* lB) {
#pragma unroll
    for (int kk = 0; kk < BK; kk += 32) {
      F16Frag af[2], bf[4];
      // A 16x32 f16 frag: lane -> row M=l16; halves[0..7]=K0..K0+7,
      // halves[8..15]=K0+16..K0+23 with K0 = kk + lhalf*8
#pragma unroll
      for (int t = 0; t < 2; ++t) {
        int row = wm * 32 + t * 16 + l16;
        af[t].h[0] = *(const v8h*)&lA[row * LDSW + kk + lhalf * 8];
        af[t].h[1] = *(const v8h*)&lA[row * LDSW + kk + lhalf * 8 + 16];
      }
      // B 32x16 f16 frag: lane -> col N=l16; same K order as A (K0..K0+7, K0+16..K0+23)
#pragma unroll
      for (int t = 0; t < 4; ++t) {
        int wrow = wn * 64 + t * 16 + l16;
        bf[t].h[0] = *(const v8h*)&lB[wrow * LDSW + kk + lhalf * 8];
        bf[t].h[1] = *(const v8h*)&lB[wrow * LDSW + kk + lhalf * 8 + 16];
      }
#pragma unroll
      for (int i = 0; i < 2; ++i)
#pragma unroll
        for (int j = 0; j < 4; ++j)
          acc[i][j] = __builtin_amdgcn_wmma_f32_16x16x32_f16(
              false, af[i].v, false, bf[j].v,
              (short)0, acc[i][j], false, false);
    }
  };

#if USE_ASYNC_LDS
  // ---- async global->LDS, 3-deep pipeline (2 tiles in flight) ----
  __shared__ _Float16 ldsA[3][128 * LDSW];   // 55 KB
  __shared__ _Float16 ldsB[3][128 * LDSW];   // 55 KB

  auto issue_tile = [&](int k0, int buf) {
#pragma unroll
    for (int i = 0; i < 4; ++i) {
      int g   = i * 256 + tid;         // 0..1023 chunk id
      int row = g >> 3;                // 0..127
      int c8  = (g & 7) << 3;          // 0,8,...,56 halves
      async_cp_b128(&A[(size_t)(m0 + row) * K + k0 + c8], &ldsA[buf][row * LDSW + c8]);
      async_cp_b128(&W[(size_t)(n0 + row) * K + k0 + c8], &ldsB[buf][row * LDSW + c8]);
    }
  };

  const int ntiles = K / BK;           // 49 (layer1) / 98 (layer2), >= 3
  issue_tile(0, 0);
  issue_tile(BK, 1);

  int cur = 0;
  // steady state: branch-free body (issue 8 copies, wait<=16, barrier,
  // 16 WMMAs, barrier)
  for (int it = 0; it + 2 < ntiles; ++it) {
    int nb = cur + 2; if (nb >= 3) nb -= 3;
    // trailing barrier of previous iteration guarantees buf nb is reusable
    issue_tile((it + 2) * BK, nb);
    __builtin_amdgcn_s_wait_asynccnt(16);  // own tile-it copies done
    __syncthreads();                       // everyone's buf[cur] visible
    compute_tile(ldsA[cur], ldsB[cur]);
    __syncthreads();                       // done reading buf[cur]
    cur = (cur == 2) ? 0 : cur + 1;
  }
  // drain: second-to-last tile
  __builtin_amdgcn_s_wait_asynccnt(8);
  __syncthreads();
  compute_tile(ldsA[cur], ldsB[cur]);
  __syncthreads();
  cur = (cur == 2) ? 0 : cur + 1;
  // drain: last tile
  __builtin_amdgcn_s_wait_asynccnt(0);
  __syncthreads();
  compute_tile(ldsA[cur], ldsB[cur]);
  __syncthreads();
#else
  // ---- fallback: register-pipelined loads, single LDS buffer ----
  __shared__ _Float16 ldsA[128 * LDSW];
  __shared__ _Float16 ldsB[128 * LDSW];
  v8h ra[4], rb[4];

  auto load_regs = [&](int k0) {
#pragma unroll
    for (int i = 0; i < 4; ++i) {
      int g   = i * 256 + tid;
      int row = g >> 3;
      int c8  = (g & 7) << 3;
      ra[i] = *(const v8h*)&A[(size_t)(m0 + row) * K + k0 + c8];
      rb[i] = *(const v8h*)&W[(size_t)(n0 + row) * K + k0 + c8];
    }
  };

  load_regs(0);
  for (int k0 = 0; k0 < K; k0 += BK) {
#pragma unroll
    for (int i = 0; i < 4; ++i) {
      int g   = i * 256 + tid;
      int row = g >> 3;
      int c8  = (g & 7) << 3;
      *(v8h*)&ldsA[row * LDSW + c8] = ra[i];
      *(v8h*)&ldsB[row * LDSW + c8] = rb[i];
    }
    __syncthreads();
    if (k0 + BK < K) load_regs(k0 + BK);  // overlap with WMMA below
    compute_tile(ldsA, ldsB);
    __syncthreads();
  }
#endif

  // ---- epilogue: threshold + store ----
  // C 16x16 f32 layout: VGPR r -> M = r + lhalf*8, N = l16
#if USE_ASYNC_LDS
  if (STORE_F16) {
    // staging LDS is dead after the final trailing barrier: reuse ldsA as a
    // 128x128 f16 tile (row stride OSTR=136 halves, conflict-free), then
    // write out coalesced b128 rows.
    _Float16* lout = &ldsA[0][0];      // needs 128*136 halves <= 3*9216  OK
#pragma unroll
    for (int i = 0; i < 2; ++i)
#pragma unroll
      for (int j = 0; j < 4; ++j)
#pragma unroll
        for (int r = 0; r < 8; ++r) {
          int row = wm * 32 + i * 16 + lhalf * 8 + r;
          int col = wn * 64 + j * 16 + l16;
          lout[row * OSTR + col] =
              (acc[i][j][r] >= 1.0f) ? (_Float16)1.0f : (_Float16)0.0f;
        }
    __syncthreads();
#pragma unroll
    for (int i = 0; i < 8; ++i) {
      int g   = i * 256 + tid;         // 0..2047
      int row = g >> 4;                // 0..127
      int c8  = (g & 15) << 3;         // 0..120 halves
      *(v8h*)&((_Float16*)Cout)[(size_t)(m0 + row) * N + n0 + c8] =
          *(const v8h*)&lout[row * OSTR + c8];
    }
    return;
  }
#endif
#pragma unroll
  for (int i = 0; i < 2; ++i) {
#pragma unroll
    for (int j = 0; j < 4; ++j) {
      int col = n0 + wn * 64 + j * 16 + l16;
#pragma unroll
      for (int r = 0; r < 8; ++r) {
        int row = m0 + wm * 32 + i * 16 + lhalf * 8 + r;
        float v = acc[i][j][r];
        if (STORE_F16) {
          ((_Float16*)Cout)[(size_t)row * N + col] =
              (v >= 1.0f) ? (_Float16)1.0f : (_Float16)0.0f;
        } else {
          if (col < Nstore)
            ((float*)Cout)[(size_t)row * Nstore + col] = (v >= 1.0f) ? 1.0f : 0.0f;
        }
      }
    }
  }
}

// ---------------------------------------------------------------------------
extern "C" void kernel_launch(void* const* d_in, const int* in_sizes, int n_in,
                              void* d_out, int out_size, void* d_ws, size_t ws_size,
                              hipStream_t stream) {
  (void)in_sizes; (void)n_in; (void)out_size; (void)ws_size;
  const float* x  = (const float*)d_in[0];   // [B][IN]   0/1 spikes
  const float* W1 = (const float*)d_in[1];   // [FEAT][IN]
  const float* W2 = (const float*)d_in[2];   // [OUT][FEAT]
  float* out = (float*)d_out;                // [B][OUT]

  const int B = 4096, IN = 3136, FEAT = 6272, OUT = 500, OUTP = 512;

  char* ws = (char*)d_ws;
  size_t off = 0;
  auto salloc = [&](size_t bytes) {
    void* p = ws + off;
    off = (off + bytes + 255) & ~(size_t)255;
    return p;
  };
  _Float16* xh  = (_Float16*)salloc((size_t)B * IN * 2);      // 25.7 MB
  _Float16* w1h = (_Float16*)salloc((size_t)FEAT * IN * 2);   // 39.3 MB
  _Float16* w2h = (_Float16*)salloc((size_t)OUTP * FEAT * 2); //  6.4 MB
  _Float16* s1h = (_Float16*)salloc((size_t)B * FEAT * 2);    // 51.4 MB

  // stage fp32 -> f16
  {
    int n4 = (B * IN) >> 2;
    cvt_f32_to_f16<<<(n4 + 255) / 256, 256, 0, stream>>>(x, xh, n4);
  }
  {
    int n4 = (FEAT * IN) >> 2;
    cvt_f32_to_f16<<<(n4 + 255) / 256, 256, 0, stream>>>(W1, w1h, n4);
  }
  {
    int n4 = (OUTP * FEAT) >> 2;
    cvt_pad_f32_to_f16<<<(n4 + 255) / 256, 256, 0, stream>>>(W2, w2h, FEAT, OUT, OUTP);
  }

  // layer 1: s1 = (x @ W1^T >= 1)  -> f16 0/1 [B][FEAT]
  {
    dim3 grid(B / 128, FEAT / 128);  // 32 x 49
    snn_gemm_nt<true><<<grid, 256, 0, stream>>>(xh, w1h, (void*)s1h, B, FEAT, IN, FEAT);
  }
  // layer 2: s2 = (s1 @ W2^T >= 1) -> f32 [B][500]
  {
    dim3 grid(B / 128, OUTP / 128);  // 32 x 4
    snn_gemm_nt<false><<<grid, 256, 0, stream>>>(s1h, w2h, (void*)out, B, OUTP, FEAT, OUT);
  }
}
